// PositionFusion_78288663872048
// MI455X (gfx1250) — matrix operand order repaired and checked by reading, not compiled, not run on hardware
//
#include <hip/hip_runtime.h>
#include <hip/hip_bf16.h>

typedef __attribute__((ext_vector_type(16))) _Float16 v16h;
typedef __attribute__((ext_vector_type(8)))  _Float16 h8;
typedef __attribute__((ext_vector_type(8)))  float    v8f;

#define BATCH 2
#define NPTS  8192
#define CIN   128
#define KNN_K 21
#define EPSBN 1e-5f

// ---------------------------------------------------------------------------
// Kernel 0: f [B,C,N] f32  ->  fT [B,N,C] f16   (contiguous per-point channel
// rows so the gather in the fused kernel is one 16B vector load per 8 chans)
// ---------------------------------------------------------------------------
__global__ __launch_bounds__(256) void transpose_kernel(
    const float* __restrict__ f, _Float16* __restrict__ fT) {
  int t = blockIdx.x * 256 + threadIdx.x;   // t = b*N + n
  int n = t & (NPTS - 1);
  int b = t >> 13;
  const float* fb = f + (size_t)b * CIN * NPTS;
  _Float16* o = fT + (size_t)t * CIN;
#pragma unroll 4
  for (int c = 0; c < CIN; ++c)
    o[c] = (_Float16)fb[(size_t)c * NPTS + n];
}

// ---------------------------------------------------------------------------
// Kernel 1: brute-force KNN (self included), top-21 per query.
// dist = |q|^2 + |c|^2 - 2 q.c  (replicates the reference formula).
// Register-resident sorted list via unrolled compare-swap insertion.
// ---------------------------------------------------------------------------
__global__ __launch_bounds__(256) void knn_kernel(
    const float* __restrict__ xyz, int* __restrict__ idxb) {
  __shared__ float cx[256], cy[256], cz[256], cd[256];
  const int b = blockIdx.y;
  const int q = blockIdx.x * 256 + threadIdx.x;
  const float* xb = xyz + (size_t)b * NPTS * 3;

  const float qx = xb[q * 3 + 0], qy = xb[q * 3 + 1], qz = xb[q * 3 + 2];
  const float qd = qx * qx + qy * qy + qz * qz;

  float bd[KNN_K];
  int   bi[KNN_K];
#pragma unroll
  for (int j = 0; j < KNN_K; ++j) { bd[j] = 3.4e38f; bi[j] = 0x7fffffff; }

  for (int c0 = 0; c0 < NPTS; c0 += 256) {
    __syncthreads();
    {
      const int ci = c0 + threadIdx.x;
      float x = xb[ci * 3 + 0], y = xb[ci * 3 + 1], z = xb[ci * 3 + 2];
      cx[threadIdx.x] = x; cy[threadIdx.x] = y; cz[threadIdx.x] = z;
      cd[threadIdx.x] = x * x + y * y + z * z;
    }
    __syncthreads();
    for (int jj = 0; jj < 256; ++jj) {
      float d = qd + cd[jj] - 2.0f * (qx * cx[jj] + qy * cy[jj] + qz * cz[jj]);
      int   id = c0 + jj;
      if (d < bd[KNN_K - 1] || (d == bd[KNN_K - 1] && id < bi[KNN_K - 1])) {
        float cv = d; int ci2 = id;
#pragma unroll
        for (int j = 0; j < KNN_K; ++j) {
          bool sw = (cv < bd[j]) || (cv == bd[j] && ci2 < bi[j]);
          if (sw) { float td = bd[j]; int ti = bi[j];
                    bd[j] = cv; bi[j] = ci2; cv = td; ci2 = ti; }
        }
      }
    }
  }
  int* o = idxb + ((size_t)b * NPTS + q) * KNN_K;
#pragma unroll
  for (int j = 0; j < KNN_K; ++j) o[j] = bi[j];
}

// ---------------------------------------------------------------------------
// Kernel 2: fused gather + (geo conv, feat conv) via v_wmma_f32_16x16x32_f16
//           + BN + ReLU + max over neighbors.
// Workgroup: 256 threads (8 waves) -> 16 points. 21 neighbors padded to 32
// slots (pads duplicate the self/center column -> max unchanged), giving two
// 16-sample WMMA N-tiles per point. Wave w owns output rows [16w,16w+16) of
// BOTH convs; all weight A-fragments live in registers.
// ---------------------------------------------------------------------------
__global__ __launch_bounds__(256) void fused_kernel(
    const float* __restrict__ xyz, const _Float16* __restrict__ fT,
    const int* __restrict__ idxb,
    const float* __restrict__ Wg, const float* __restrict__ Wf,
    const float* __restrict__ g_geo, const float* __restrict__ b_geo,
    const float* __restrict__ m_geo, const float* __restrict__ v_geo,
    const float* __restrict__ g_ft,  const float* __restrict__ b_ft,
    const float* __restrict__ m_ft,  const float* __restrict__ v_ft,
    float* __restrict__ out) {
  // Sample-major B-tiles in LDS (f16): Gt = feat (256 ch), Gg = geo (32 ch, padded)
  __shared__ _Float16 Gt[16][264];   // stride 264 -> 528B rows, conflict-friendly
  __shared__ _Float16 Gg[16][40];    // stride 40  -> 80B rows

  const int b   = blockIdx.y;
  const int p0  = blockIdx.x * 16;
  const int tid = threadIdx.x;
  const int lane = tid & 31, w = tid >> 5;
  const int col = lane & 15, hi = lane >> 4;
  const int arow = 16 * w + col;          // A-matrix row (M) for this lane

  union BF { v16h h; unsigned int u[8]; };

  // ---- Preload A fragments (ISA 16-bit A 16x32 layout) -------------------
  // dword d of lane (row=arow, hi) holds K = k0 + (d&4?16:0) + 8*hi + 2*(d&3)
  v16h Af[8];
#pragma unroll
  for (int kk = 0; kk < 8; ++kk) {
    v16h a;
#pragma unroll
    for (int d = 0; d < 8; ++d) {
      int k = kk * 32 + ((d & 4) ? 16 : 0) + 8 * hi + (d & 3) * 2;
      a[2 * d + 0] = (_Float16)Wf[arow * 256 + k + 0];
      a[2 * d + 1] = (_Float16)Wf[arow * 256 + k + 1];
    }
    Af[kk] = a;
  }
  v16h Ag;
  {
    v16h a;
#pragma unroll
    for (int d = 0; d < 8; ++d) {
      int k = ((d & 4) ? 16 : 0) + 8 * hi + (d & 3) * 2;
      a[2 * d + 0] = (k + 0 < 6) ? (_Float16)Wg[arow * 6 + k + 0] : (_Float16)0.0f;
      a[2 * d + 1] = (k + 1 < 6) ? (_Float16)Wg[arow * 6 + k + 1] : (_Float16)0.0f;
    }
    Ag = a;
  }

  // ---- Per-lane BN constants for the 8 C/D rows it holds -----------------
  // C/D layout: VGPR v holds row M = v + 8*hi, col = lane&15.
  float sg[8], hg[8], sf[8], hf[8];
#pragma unroll
  for (int v = 0; v < 8; ++v) {
    int ch = 16 * w + v + 8 * hi;
    float iv = g_geo[ch] * rsqrtf(v_geo[ch] + EPSBN);
    sg[v] = iv; hg[v] = b_geo[ch] - m_geo[ch] * iv;
    float jv = g_ft[ch] * rsqrtf(v_ft[ch] + EPSBN);
    sf[v] = jv; hf[v] = b_ft[ch] - m_ft[ch] * jv;
  }

  // zero-pad geo channels 6..31 once (never rewritten)
  if (tid < 16)
    for (int k = 6; k < 32; ++k) Gg[tid][k] = (_Float16)0.0f;

  const _Float16* fTb   = fT + (size_t)b * NPTS * CIN;
  const int*      idxB  = idxb + ((size_t)b * NPTS + p0) * KNN_K;
  const float*    xyzb  = xyz + (size_t)b * NPTS * 3;

  float mg[8], mf[8];

  for (int t = 0; t < 32; ++t) {           // 16 points x 2 slot-tiles
    __syncthreads();                       // prev tile's LDS reads complete
    const int pl = t >> 1;                 // local point
    const int p  = p0 + pl;
    const int jb = (t & 1) * 16;           // slot base (0 or 16)

    // ---- stage feat B-tile: 16 threads per sample, 8 channels per thread --
    {
      const int s  = tid >> 4;             // sample column 0..15
      const int c8 = tid & 15;             // channel chunk
      const int j  = jb + s;
      const int nb = (j < KNN_K) ? idxB[pl * KNN_K + j] : p;
      h8 hc = *(const h8*)(fTb + (size_t)p  * CIN + c8 * 8);
      h8 hn = *(const h8*)(fTb + (size_t)nb * CIN + c8 * 8);
#pragma unroll
      for (int i = 0; i < 8; ++i) {
        Gt[s][c8 * 8 + i]       = hc[i];                      // center half
        Gt[s][128 + c8 * 8 + i] = (_Float16)(hn[i] - hc[i]);  // diff half
      }
      // ---- stage geo B-tile (first 16 threads) ----
      if (tid < 16) {
        float cxx = xyzb[p * 3 + 0], cyy = xyzb[p * 3 + 1], czz = xyzb[p * 3 + 2];
        float nxx = xyzb[nb * 3 + 0], nyy = xyzb[nb * 3 + 1], nzz = xyzb[nb * 3 + 2];
        Gg[s][0] = (_Float16)cxx; Gg[s][1] = (_Float16)cyy; Gg[s][2] = (_Float16)czz;
        Gg[s][3] = (_Float16)(nxx - cxx);
        Gg[s][4] = (_Float16)(nyy - cyy);
        Gg[s][5] = (_Float16)(nzz - czz);
      }
    }
    __syncthreads();

    // ---- WMMA: B-frag dword d holds K = (d&4?16:0) + 8*hi + 2*(d&3) (pairs), col = lane&15 --
    BF bg;
#pragma unroll
    for (int d = 0; d < 8; ++d)
      bg.u[d] = *(const unsigned int*)(&Gg[col][0] + ((d & 4) ? 16 : 0) + 8 * hi + 2 * (d & 3));

    v8f zero = {};
    v8f cg = __builtin_amdgcn_wmma_f32_16x16x32_f16(
        false, Ag, false, bg.h, (short)0, zero, false, false);

    v8f cf = {};
#pragma unroll
    for (int kk = 0; kk < 8; ++kk) {
      BF bf;
#pragma unroll
      for (int d = 0; d < 8; ++d)
        bf.u[d] = *(const unsigned int*)(&Gt[col][0] + kk * 32 + ((d & 4) ? 16 : 0) + 8 * hi + 2 * (d & 3));
      cf = __builtin_amdgcn_wmma_f32_16x16x32_f16(
          false, Af[kk], false, bf.h, (short)0, cf, false, false);
    }

    // ---- BN + ReLU, running max over the point's two slot-tiles ----------
#pragma unroll
    for (int v = 0; v < 8; ++v) {
      float yg = fmaxf(0.0f, cg[v] * sg[v] + hg[v]);
      float yf = fmaxf(0.0f, cf[v] * sf[v] + hf[v]);
      if ((t & 1) == 0) { mg[v] = yg; mf[v] = yf; }
      else              { mg[v] = fmaxf(mg[v], yg); mf[v] = fmaxf(mf[v], yf); }
    }

    // ---- after 2nd slot-tile: reduce max over 16 columns, store ----------
    if (t & 1) {
#pragma unroll
      for (int v = 0; v < 8; ++v) {
        float m1 = mg[v], m2 = mf[v];
#pragma unroll
        for (int off = 1; off < 16; off <<= 1) {
          m1 = fmaxf(m1, __shfl_xor(m1, off, 32));
          m2 = fmaxf(m2, __shfl_xor(m2, off, 32));
        }
        if (col == 0) {
          int ch = 16 * w + v + 8 * hi;
          out[((size_t)b * 256 + ch)       * NPTS + p] = m1;  // geo block
          out[((size_t)b * 256 + 128 + ch) * NPTS + p] = m2;  // feat block
        }
      }
    }
  }
}

// ---------------------------------------------------------------------------
extern "C" void kernel_launch(void* const* d_in, const int* in_sizes, int n_in,
                              void* d_out, int out_size, void* d_ws, size_t ws_size,
                              hipStream_t stream) {
  (void)in_sizes; (void)n_in; (void)out_size; (void)ws_size;
  const float* xyz   = (const float*)d_in[0];
  const float* f     = (const float*)d_in[1];
  const float* W_geo = (const float*)d_in[2];
  const float* g_geo = (const float*)d_in[3];
  const float* b_geo = (const float*)d_in[4];
  const float* m_geo = (const float*)d_in[5];
  const float* v_geo = (const float*)d_in[6];
  const float* W_ft  = (const float*)d_in[7];
  const float* g_ft  = (const float*)d_in[8];
  const float* b_ft  = (const float*)d_in[9];
  const float* m_ft  = (const float*)d_in[10];
  const float* v_ft  = (const float*)d_in[11];
  float* out = (float*)d_out;

  _Float16* fT = (_Float16*)d_ws;                                  // 4 MB
  int* idxb = (int*)((char*)d_ws + (size_t)BATCH * NPTS * CIN * 2);// 1.4 MB

  transpose_kernel<<<(BATCH * NPTS) / 256, 256, 0, stream>>>(f, fT);
  knn_kernel<<<dim3(NPTS / 256, BATCH), 256, 0, stream>>>(xyz, idxb);
  fused_kernel<<<dim3(NPTS / 16, BATCH), 256, 0, stream>>>(
      xyz, fT, idxb, W_geo, W_ft,
      g_geo, b_geo, m_geo, v_geo,
      g_ft, b_ft, m_ft, v_ft, out);
}
